// MultiHeadAttention_52226802319567
// MI455X (gfx1250) — compile-verified, matrix operand order checked by reading, not run on hardware
//
#include <hip/hip_runtime.h>
#include <hip/hip_bf16.h>

// MHA: B=4, S=2048, E=1024, H=16, D=64.  bf16 WMMA (16x16x32), f32 accum.
#define E_DIM   1024
#define NHEADS  16
#define HDIM    64
#define BATCH   4
#define SEQ     2048

typedef __bf16 bf16;
typedef __bf16 v16bf __attribute__((ext_vector_type(16)));
typedef __bf16 v4bf  __attribute__((ext_vector_type(4)));
typedef float  v8f   __attribute__((ext_vector_type(8)));

#define WMMA_BF16(a, b, c) \
  __builtin_amdgcn_wmma_f32_16x16x32_bf16(false, (a), false, (b), (short)0, (c), false, false)

// Load one 16x32 bf16 fragment (A or B^T layout) from an LDS tile.
// CDNA5 ISA 7.12.2: lanes 0-15 hold row r=lane, K 0-7 then 16-23;
// lanes 16-31 hold row r=lane-16, K 8-15 then 24-31.  Two b128 reads.
__device__ inline v16bf ld_frag_lds(const bf16* rowbase, int stride_el) {
  int lane = threadIdx.x & 31;
  const bf16* q = rowbase + (lane & 15) * stride_el + ((lane >> 4) << 3);
  union { v16bf v; uint4 u[2]; } t;
  t.u[0] = *(const uint4*)(q);
  t.u[1] = *(const uint4*)(q + 16);
  return t.v;
}

// ---------------------------------------------------------------- fp32->bf16
__global__ void cvt_f32_bf16_kernel(const float* __restrict__ src,
                                    bf16* __restrict__ dst, int n4) {
  int i = blockIdx.x * 256 + threadIdx.x;
  if (i < n4) {
    float4 f = ((const float4*)src)[i];
    v4bf o = { (bf16)f.x, (bf16)f.y, (bf16)f.z, (bf16)f.w };
    ((v4bf*)dst)[i] = o;
  }
}

// ------------------------------------------------------------------- GEMMs
// C[m,n] = sum_k A[m,k] * W[n,k] + bias[n]   (A:[M,1024], W:[N,1024], bf16)
// Block tile 128x128, 8 waves, wave tile 32x64 (2x4 WMMA tiles).
// K-tile 32, double-buffered LDS, one barrier per tile; global loads for
// tile t+1 overlap the WMMA of tile t.
// MODE 0: scatter (qkv) epilogue -> head-major bf16 Q/K/V [B,H,S,D]
//         (Q pre-scaled by log2(e)/sqrt(D) so softmax can run in exp2 domain)
// MODE 1: plain fp32 store to Out [M, E_DIM]
template <int MODE>
__global__ __launch_bounds__(256) void gemm_bf16_kernel(
    const bf16* __restrict__ A, const bf16* __restrict__ W,
    const float* __restrict__ bias,
    bf16* __restrict__ Qb, bf16* __restrict__ Kb, bf16* __restrict__ Vb,
    float* __restrict__ Out) {
  const int Kdim = 1024, NT = Kdim / 32;
  __shared__ __align__(16) bf16 As[2][128][40];  // 128 rows x 32 K, +8 pad
  __shared__ __align__(16) bf16 Bs[2][128][40];
  const int mbase = blockIdx.x * 128;
  const int nbase = blockIdx.y * 128;
  const int tid = threadIdx.x, lane = tid & 31, wave = tid >> 5;
  const int wr = (wave >> 1) * 32;   // 4 row-waves * 32 rows
  const int wc = (wave & 1) * 64;    // 2 col-waves * 64 cols
  const int lr = tid >> 1;           // staging: row 0..127
  const int lc = (tid & 1) * 16;     // staging: 16-element half

  const bf16* Arow = A + (size_t)(mbase + lr) * Kdim + lc;
  const bf16* Wrow = W + (size_t)(nbase + lr) * Kdim + lc;

  v8f acc[2][4] = {};
  uint4 rA[2], rB[2];

  // prologue: fetch tile 0 into registers
  rA[0] = *(const uint4*)(Arow);      rA[1] = *(const uint4*)(Arow + 8);
  rB[0] = *(const uint4*)(Wrow);      rB[1] = *(const uint4*)(Wrow + 8);

  for (int kt = 0; kt < NT; ++kt) {
    const int b = kt & 1;
    // commit staged registers to LDS buffer b
    *(uint4*)&As[b][lr][lc]     = rA[0];
    *(uint4*)&As[b][lr][lc + 8] = rA[1];
    *(uint4*)&Bs[b][lr][lc]     = rB[0];
    *(uint4*)&Bs[b][lr][lc + 8] = rB[1];
    // issue global loads for tile kt+1 (overlaps compute below)
    if (kt + 1 < NT) {
      const bf16* ga = Arow + (kt + 1) * 32;
      const bf16* gb = Wrow + (kt + 1) * 32;
      rA[0] = *(const uint4*)(ga);    rA[1] = *(const uint4*)(ga + 8);
      rB[0] = *(const uint4*)(gb);    rB[1] = *(const uint4*)(gb + 8);
    }
    __syncthreads();

    v16bf a0 = ld_frag_lds(&As[b][wr][0], 40);
    v16bf a1 = ld_frag_lds(&As[b][wr + 16][0], 40);
#pragma unroll
    for (int j = 0; j < 4; j++) {
      v16bf bb = ld_frag_lds(&Bs[b][wc + 16 * j][0], 40);
      acc[0][j] = WMMA_BF16(a0, bb, acc[0][j]);
      acc[1][j] = WMMA_BF16(a1, bb, acc[1][j]);
    }
  }

  // Epilogue.  C/D layout: col = lane%16, row = vgpr + 8*(lane/16).
  const int lh = lane >> 4, lcol = lane & 15;
#pragma unroll
  for (int i = 0; i < 2; i++) {
#pragma unroll
    for (int j = 0; j < 4; j++) {
      int n = nbase + wc + 16 * j + lcol;
      float bv = bias[n];
#pragma unroll
      for (int v = 0; v < 8; v++) {
        int m = mbase + wr + 16 * i + v + 8 * lh;
        float val = acc[i][j][v] + bv;
        if (MODE == 0) {
          int which = n >> 10;           // 0:q 1:k 2:v
          int e = n & 1023, hh = e >> 6, dd = e & 63;
          int bb = m >> 11, ss = m & 2047;
          // fold (1/sqrt(64)) * log2(e) into Q for exp2-domain softmax
          if (which == 0) val *= 0.18033688011112042f;
          bf16* dst = (which == 0) ? Qb : ((which == 1) ? Kb : Vb);
          dst[(((size_t)(bb * NHEADS + hh)) * SEQ + ss) * HDIM + dd] = (bf16)val;
        } else {
          Out[(size_t)m * E_DIM + n] = val;
        }
      }
    }
  }
}

// --------------------------------------------------------- flash attention
// One block = 128 query rows of one (b,h).  8 waves, 16 query rows each.
// Computes S^T = K*Q^T so the probability matrix stays in registers:
// the C-layout of S^T (col = qrow = lane%16, rows = keys v+8*lh+16t)
// matches the B-fragment layout needed for O^T = V^T * P^T exactly.
// K/V tiles double-buffered in LDS; one barrier per 64-key tile.
__global__ __launch_bounds__(256) void attn_kernel(
    const bf16* __restrict__ Qb, const bf16* __restrict__ Kb,
    const bf16* __restrict__ Vb, bf16* __restrict__ Ob) {
  __shared__ __align__(16) bf16 Qs[128][72];
  __shared__ __align__(16) bf16 Ks[2][64][72];
  __shared__ __align__(16) bf16 Vt[2][64][72];   // transposed: Vt[.][d][key]

  const int tid = threadIdx.x, lane = tid & 31, wave = tid >> 5;
  const int qblk = blockIdx.x;       // 0..15
  const int bh = blockIdx.y;         // 0..63 (= b*16 + h)
  const size_t head_off = (size_t)bh * SEQ * HDIM;
  const bf16* Qg = Qb + head_off + (size_t)qblk * 128 * HDIM;
  const bf16* Kg = Kb + head_off;
  const bf16* Vg = Vb + head_off;

  // Q staging: 128 rows x 64, 256 threads -> 32 elems each
  {
    const int lr = tid >> 1, lc = (tid & 1) * 32;
#pragma unroll
    for (int c = 0; c < 4; c++)
      *(uint4*)&Qs[lr][lc + c * 8] = *(const uint4*)(Qg + lr * HDIM + lc + c * 8);
  }

  const int wr = wave * 16;          // this wave's query rows (0..112)
  const int lh = lane >> 4, lcol = lane & 15;
  // per-lane online-softmax state for query row (wr + lcol), replicated x2
  float mprev = -1e30f, lsum = 0.f;
  v8f accO[4] = {};                  // O^T tiles: row d = 16t+8lh+v, col qrow

  const int kr = tid >> 2;           // K/V staging row 0..63
  const int kc = (tid & 3) * 16;     // 16-element chunk
  const bf16* Krow = Kg + (size_t)kr * HDIM + kc;
  const bf16* Vrow = Vg + (size_t)kr * HDIM + kc;

  uint4 kregs[2], vregs[2];
  // prologue: fetch key-tile 0
  kregs[0] = *(const uint4*)(Krow);     kregs[1] = *(const uint4*)(Krow + 8);
  vregs[0] = *(const uint4*)(Vrow);     vregs[1] = *(const uint4*)(Vrow + 8);

  for (int j = 0; j < SEQ; j += 64) {
    const int b = (j >> 6) & 1;
    // commit staged K tile + transposed V tile into LDS buffer b
    *(uint4*)&Ks[b][kr][kc]     = kregs[0];
    *(uint4*)&Ks[b][kr][kc + 8] = kregs[1];
#pragma unroll
    for (int c = 0; c < 2; c++) {
      union { uint4 u; bf16 e[8]; } t;
      t.u = vregs[c];
#pragma unroll
      for (int q = 0; q < 8; q++) Vt[b][kc + c * 8 + q][kr] = t.e[q];
    }
    // issue global loads for the next key tile (overlaps compute below)
    if (j + 64 < SEQ) {
      const bf16* kg = Krow + (size_t)(j + 64) * HDIM;
      const bf16* vg = Vrow + (size_t)(j + 64) * HDIM;
      kregs[0] = *(const uint4*)(kg);   kregs[1] = *(const uint4*)(kg + 8);
      vregs[0] = *(const uint4*)(vg);   vregs[1] = *(const uint4*)(vg + 8);
    }
    __syncthreads();

    // S^T = K * Q^T : tile t covers keys 16t..16t+15, cols = 16 q-rows
    v8f sc[4] = {};
#pragma unroll
    for (int kk = 0; kk < 64; kk += 32) {
      v16bf bq = ld_frag_lds(&Qs[wr][kk], 72);
#pragma unroll
      for (int t = 0; t < 4; t++) {
        v16bf ak = ld_frag_lds(&Ks[b][16 * t][kk], 72);
        sc[t] = WMMA_BF16(ak, bq, sc[t]);
      }
    }

    // online softmax in exp2 domain; each lane owns 32 key-scores of one qrow
    float mx = sc[0][0];
#pragma unroll
    for (int t = 0; t < 4; t++)
#pragma unroll
      for (int v = 0; v < 8; v++) mx = fmaxf(mx, sc[t][v]);
    mx = fmaxf(mx, __shfl_xor(mx, 16, 32));   // combine the two key-halves
    float mn = fmaxf(mprev, mx);
    float al = exp2f(mprev - mn);
    float rs = 0.f;
    v16bf pf[2];  // P^T B-fragments: pf[0]=keys 0..31, pf[1]=keys 32..63
#pragma unroll
    for (int t = 0; t < 4; t++)
#pragma unroll
      for (int v = 0; v < 8; v++) {
        float pv = exp2f(sc[t][v] - mn);
        rs += pv;
        pf[t >> 1][(t & 1) * 8 + v] = (bf16)pv;
      }
    rs += __shfl_xor(rs, 16, 32);
    lsum = lsum * al + rs;
    mprev = mn;
#pragma unroll
    for (int t = 0; t < 4; t++)
#pragma unroll
      for (int v = 0; v < 8; v++) accO[t][v] *= al;

    // O^T += V^T * P^T  (A = Vt rows d, K-dim = 64 keys; B = pf from regs)
#pragma unroll
    for (int half = 0; half < 2; half++) {
#pragma unroll
      for (int t = 0; t < 4; t++) {
        v16bf av = ld_frag_lds(&Vt[b][16 * t][half * 32], 72);
        accO[t] = WMMA_BF16(av, pf[half], accO[t]);
      }
    }
  }

  // normalize + write attn output in [B,S,E] bf16 layout.
  // Each lane holds 8 consecutive d-values per tile -> one b128 store each.
  const int bb = bh >> 4, hh = bh & 15;
  const float inv = 1.0f / lsum;
  const int s = qblk * 128 + wr + lcol;
  bf16* orow = Ob + ((size_t)(bb * SEQ + s)) * E_DIM + hh * HDIM + 8 * lh;
#pragma unroll
  for (int t = 0; t < 4; t++) {
    union { uint4 u; bf16 b[8]; } o;
#pragma unroll
    for (int v = 0; v < 8; v++) o.b[v] = (bf16)(accO[t][v] * inv);
    *(uint4*)(orow + 16 * t) = o.u;
  }
}

// ------------------------------------------------------------------ launch
extern "C" void kernel_launch(void* const* d_in, const int* in_sizes, int n_in,
                              void* d_out, int out_size, void* d_ws, size_t ws_size,
                              hipStream_t stream) {
  (void)in_sizes; (void)n_in; (void)out_size; (void)ws_size;
  const float* x     = (const float*)d_in[0];
  const float* W_qkv = (const float*)d_in[1];
  const float* b_qkv = (const float*)d_in[2];
  const float* W_out = (const float*)d_in[3];
  const float* b_out = (const float*)d_in[4];

  char* ws = (char*)d_ws;
  bf16* xb  = (bf16*)(ws);                     // [8192,1024]      16 MiB
  bf16* wqb = (bf16*)(ws + 16777216);          // [3072,1024]       6 MiB
  bf16* wob = (bf16*)(ws + 23068672);          // [1024,1024]       2 MiB
  bf16* Qb  = (bf16*)(ws + 25165824);          // [B,H,S,D]        16 MiB
  bf16* Kb  = (bf16*)(ws + 41943040);          // [B,H,S,D]        16 MiB
  bf16* Vb  = (bf16*)(ws + 58720256);          // [B,H,S,D]        16 MiB
  bf16* Ob  = (bf16*)(ws + 75497472);          // [B,S,E]          16 MiB

  cvt_f32_bf16_kernel<<<8192, 256, 0, stream>>>(x, xb, 8192 * 1024 / 4);
  cvt_f32_bf16_kernel<<<3072, 256, 0, stream>>>(W_qkv, wqb, 3072 * 1024 / 4);
  cvt_f32_bf16_kernel<<<1024, 256, 0, stream>>>(W_out, wob, 1024 * 1024 / 4);

  // QKV projection: M=8192, N=3072, K=1024
  gemm_bf16_kernel<0><<<dim3(64, 24), 256, 0, stream>>>(
      xb, wqb, b_qkv, Qb, Kb, Vb, nullptr);

  // flash attention: 16 query blocks x 64 (b,h) pairs
  attn_kernel<<<dim3(16, 64), 256, 0, stream>>>(Qb, Kb, Vb, Ob);

  // output projection: M=8192, N=1024, K=1024, fp32 out
  gemm_bf16_kernel<1><<<dim3(64, 8), 256, 0, stream>>>(
      Ob, wob, b_out, nullptr, nullptr, nullptr, (float*)d_out);
}
